// SelfAttention_45801531245209
// MI455X (gfx1250) — matrix operand order repaired and checked by reading, not compiled, not run on hardware
//
#include <hip/hip_runtime.h>

constexpr int kB = 4, kN = 4096, kC = 256, kD = 64;

typedef __attribute__((ext_vector_type(16))) __bf16 bf16x16;
typedef __attribute__((ext_vector_type(8)))  float  f32x8;

struct alignas(16) U4 { unsigned int x, y, z, w; };
struct alignas(16) F4 { float x, y, z, w; };

union Frag { bf16x16 f; U4 u[2]; };

static __device__ __forceinline__ unsigned short f2bf(float x) {
  unsigned int u = __float_as_uint(x);
  u += 0x7FFFu + ((u >> 16) & 1u);          // round to nearest even
  return (unsigned short)(u >> 16);
}

static __device__ __forceinline__ f32x8 wmma_bf16(const Frag& a, const Frag& b, f32x8 c) {
  return __builtin_amdgcn_wmma_f32_16x16x32_bf16(false, a.f, false, b.f, (short)0, c,
                                                 false, false);
}

// Async global -> LDS copy of one 16B chunk per lane (ASYNCcnt-tracked).
static __device__ __forceinline__ void async_b128(const unsigned short* gsrc,
                                                  unsigned short* ldst) {
  unsigned int       la = (unsigned int)(size_t)ldst;       // low 32b = LDS offset
  unsigned long long ga = (unsigned long long)(size_t)gsrc;
  asm volatile("global_load_async_to_lds_b128 %0, %1, off" :: "v"(la), "v"(ga)
               : "memory");
}

// ---------------------------------------------------------------------------
// Weight convert + transpose: dst[col*R + row] = bf16(src[row*Cc + col])
// ---------------------------------------------------------------------------
__global__ void sa_wconv_kernel(const float* __restrict__ src,
                                unsigned short* __restrict__ dst, int R, int Cc) {
  int i = blockIdx.x * blockDim.x + threadIdx.x;
  if (i >= R * Cc) return;
  int r = i / Cc, c = i - r * Cc;
  dst[c * R + r] = f2bf(src[i]);
}

// ---------------------------------------------------------------------------
// Projections: h = pWh+bh, l = pWl+bl (row-major bf16 [tok][64]),
//              g = pWg+bg stored transposed bf16 [B][C][N].
// ---------------------------------------------------------------------------
__global__ __launch_bounds__(128)
void sa_proj_kernel(const float* __restrict__ p,
                    const unsigned short* __restrict__ WhT,
                    const unsigned short* __restrict__ WlT,
                    const unsigned short* __restrict__ WgT,
                    const float* __restrict__ bh,
                    const float* __restrict__ bl,
                    const float* __restrict__ bg,
                    unsigned short* __restrict__ h16,
                    unsigned short* __restrict__ l16,
                    unsigned short* __restrict__ g16) {
  __shared__ unsigned short sp[64 * kC];      // 32KB: 64 p-rows as bf16
  const int tid     = threadIdx.x;
  const int lane    = tid & 31;
  const int wave    = tid >> 5;
  const int tokBase = blockIdx.x * 64;
  const int lcol    = lane & 15;
  const int hi      = (lane >> 4) & 1;

  // ---- stage p tile into LDS as bf16 ----
  const F4* p4 = (const F4*)p + tokBase * (kC / 4);
  for (int i = tid; i < 64 * kC / 4; i += 128) {
    F4 v = p4[i];
    unsigned int w0 = f2bf(v.x) | ((unsigned int)f2bf(v.y) << 16);
    unsigned int w1 = f2bf(v.z) | ((unsigned int)f2bf(v.w) << 16);
    *(unsigned int*)&sp[i * 4 + 0] = w0;
    *(unsigned int*)&sp[i * 4 + 2] = w1;
  }
  __syncthreads();

  const int row16 = wave * 16;
  const int aRow  = row16 + lcol;

  for (int mat = 0; mat < 3; ++mat) {
    const unsigned short* WT = (mat == 0) ? WhT : (mat == 1) ? WlT : WgT;
    const float* bias        = (mat == 0) ? bh  : (mat == 1) ? bl  : bg;
    const int nColTiles      = (mat == 2) ? (kC / 16) : (kD / 16);
    for (int ct = 0; ct < nColTiles; ++ct) {
      const int col0 = ct * 16;
      f32x8 acc = {};
      #pragma unroll
      for (int kc = 0; kc < kC / 32; ++kc) {
        Frag a, b;
        const int abase = aRow * kC + kc * 32 + (hi ? 8 : 0);
        a.u[0] = *(const U4*)&sp[abase];
        a.u[1] = *(const U4*)&sp[abase + 16];
        const unsigned short* wrow = WT + (col0 + lcol) * kC + kc * 32 + (hi ? 8 : 0);
        b.u[0] = *(const U4*)wrow;
        b.u[1] = *(const U4*)(wrow + 16);
        acc = wmma_bf16(a, b, acc);
      }
      const float bv = bias[col0 + lcol];
      #pragma unroll
      for (int r = 0; r < 8; ++r) acc[r] += bv;

      if (mat < 2) {
        unsigned short* dst = (mat == 0) ? h16 : l16;
        #pragma unroll
        for (int r = 0; r < 8; ++r) {
          const int tok = tokBase + row16 + r + hi * 8;
          dst[tok * kD + col0 + lcol] = f2bf(acc[r]);
        }
      } else {
        const int ch  = col0 + lcol;
        const int tok = tokBase + row16 + hi * 8;
        const int b0  = tok >> 12;
        const int nt  = tok & (kN - 1);
        U4 q;
        q.x = f2bf(acc[0]) | ((unsigned int)f2bf(acc[1]) << 16);
        q.y = f2bf(acc[2]) | ((unsigned int)f2bf(acc[3]) << 16);
        q.z = f2bf(acc[4]) | ((unsigned int)f2bf(acc[5]) << 16);
        q.w = f2bf(acc[6]) | ((unsigned int)f2bf(acc[7]) << 16);
        *(U4*)&g16[(b0 * kC + ch) * kN + nt] = q;
      }
    }
  }
}

// ---------------------------------------------------------------------------
// Flash attention with double-buffered async-LDS staging of key tiles.
// Block = 4 waves = 64 query rows; all waves share each 32-key h/g tile.
// ---------------------------------------------------------------------------
__global__ __launch_bounds__(128)
void sa_attn_kernel(const float* __restrict__ p,
                    const unsigned short* __restrict__ h16,
                    const unsigned short* __restrict__ l16,
                    const unsigned short* __restrict__ g16,
                    float* __restrict__ out) {
  __shared__ unsigned short sH[2][32 * kD];   // 2 x 4KB  : h key tile
  __shared__ unsigned short sG[2][kC * 32];   // 2 x 16KB : g key tile (chan-major)
  __shared__ unsigned short sP[4][16 * 32];   // per-wave P transpose scratch

  const int tid  = threadIdx.x;
  const int lane = tid & 31;
  const int wave = tid >> 5;
  const int lcol = lane & 15;
  const int hi   = (lane >> 4) & 1;

  const int t0 = blockIdx.x * 64 + wave * 16;  // query tile (global token)
  const int b0 = t0 >> 12;

  // Cooperative async stage of key tile kt into buffer buf.
  auto stage = [&](int buf, int kt) {
    const int klocal = kt * 32;
    const int kglob  = (b0 << 12) + klocal;
    // h tile: 32 rows x 64 bf16, contiguous 4KB
    #pragma unroll
    for (int c = 0; c < 2; ++c) {
      const int off = (c * 128 + tid) * 8;     // ushort units, 16B chunks
      async_b128(h16 + kglob * kD + off, &sH[buf][off]);
    }
    // g tile: 256 channel rows x 32 bf16 (64B each, strided in global)
    #pragma unroll
    for (int c = 0; c < 8; ++c) {
      const int j    = c * 128 + tid;          // 16B chunk id, 4 per row
      const int row  = j >> 2;
      const int part = (j & 3) * 8;
      async_b128(g16 + (size_t)(b0 * kC + row) * kN + klocal + part,
                 &sG[buf][row * 32 + part]);
    }
  };

  // l A-fragments (16 x 64) kept in registers for the whole key loop
  Frag la[2];
  {
    const int arow = t0 + lcol;
    #pragma unroll
    for (int c = 0; c < 2; ++c) {
      const unsigned short* src = l16 + arow * kD + c * 32 + (hi ? 8 : 0);
      la[c].u[0] = *(const U4*)src;
      la[c].u[1] = *(const U4*)(src + 16);
    }
  }

  f32x8 O[16];
  #pragma unroll
  for (int t = 0; t < 16; ++t) O[t] = (f32x8){};
  float mrow[8], srow[8];
  #pragma unroll
  for (int r = 0; r < 8; ++r) { mrow[r] = -3.0e38f; srow[r] = 0.0f; }

  unsigned short* myP = sP[wave];

  stage(0, 0);

  #pragma unroll 1
  for (int kt = 0; kt < kN / 32; ++kt) {
    const int cur = kt & 1;
    asm volatile("s_wait_asynccnt 0x0" ::: "memory");
    __syncthreads();                           // tile kt resident for all waves
    if (kt + 1 < kN / 32) stage(1 - cur, kt + 1);

    // ---- scores S (16x32) = l (16x64) x h^T (64x32): 4 WMMAs ----
    f32x8 s0 = {}, s1 = {};
    #pragma unroll
    for (int c = 0; c < 2; ++c) {
      Frag hb;
      const unsigned short* src = &sH[cur][lcol * kD + c * 32 + (hi ? 8 : 0)];
      hb.u[0] = *(const U4*)src;
      hb.u[1] = *(const U4*)(src + 16);
      s0 = wmma_bf16(la[c], hb, s0);
    }
    #pragma unroll
    for (int c = 0; c < 2; ++c) {
      Frag hb;
      const unsigned short* src = &sH[cur][(16 + lcol) * kD + c * 32 + (hi ? 8 : 0)];
      hb.u[0] = *(const U4*)src;
      hb.u[1] = *(const U4*)(src + 16);
      s1 = wmma_bf16(la[c], hb, s1);
    }

    // ---- online softmax (row reductions via 16-lane butterflies) ----
    float alpha[8];
    #pragma unroll
    for (int r = 0; r < 8; ++r) {
      float v = fmaxf(s0[r], s1[r]);
      v = fmaxf(v, __shfl_xor(v, 1, 32));
      v = fmaxf(v, __shfl_xor(v, 2, 32));
      v = fmaxf(v, __shfl_xor(v, 4, 32));
      v = fmaxf(v, __shfl_xor(v, 8, 32));
      const float nm = fmaxf(mrow[r], v);
      alpha[r] = __expf(mrow[r] - nm);
      mrow[r]  = nm;
      s0[r] = __expf(s0[r] - nm);
      s1[r] = __expf(s1[r] - nm);
      float rs = s0[r] + s1[r];
      rs += __shfl_xor(rs, 1, 32);
      rs += __shfl_xor(rs, 2, 32);
      rs += __shfl_xor(rs, 4, 32);
      rs += __shfl_xor(rs, 8, 32);
      srow[r] = srow[r] * alpha[r] + rs;
    }
    #pragma unroll
    for (int t = 0; t < 16; ++t) {
      #pragma unroll
      for (int r = 0; r < 8; ++r) O[t][r] *= alpha[r];
    }

    // ---- transpose P: C-tile layout -> A-fragment layout via LDS ----
    #pragma unroll
    for (int r = 0; r < 8; ++r) {
      const int R = r + hi * 8;
      myP[R * 32 + lcol]      = f2bf(s0[r]);
      myP[R * 32 + 16 + lcol] = f2bf(s1[r]);
    }
    asm volatile("s_wait_dscnt 0x0" ::: "memory");  // wave-private RAW on LDS
    Frag pa;
    {
      const int base = lcol * 32 + (hi ? 8 : 0);
      pa.u[0] = *(const U4*)&myP[base];
      pa.u[1] = *(const U4*)&myP[base + 16];
    }

    // ---- O (16x256) += P (16x32) x g (32x256): 16 WMMAs ----
    #pragma unroll
    for (int t = 0; t < 16; ++t) {
      Frag gb;
      const unsigned short* src = &sG[cur][(t * 16 + lcol) * 32 + (hi ? 8 : 0)];
      gb.u[0] = *(const U4*)src;
      gb.u[1] = *(const U4*)(src + 16);
      O[t] = wmma_bf16(pa, gb, O[t]);
    }
  }

  // ---- epilogue: out = p + O / rowsum ----
  float rinv[8];
  #pragma unroll
  for (int r = 0; r < 8; ++r) rinv[r] = 1.0f / srow[r];
  #pragma unroll
  for (int t = 0; t < 16; ++t) {
    #pragma unroll
    for (int r = 0; r < 8; ++r) {
      const int tok = t0 + r + hi * 8;
      const int idx = tok * kC + t * 16 + lcol;
      out[idx] = p[idx] + O[t][r] * rinv[r];
    }
  }
}

// ---------------------------------------------------------------------------
extern "C" void kernel_launch(void* const* d_in, const int* in_sizes, int n_in,
                              void* d_out, int out_size, void* d_ws, size_t ws_size,
                              hipStream_t stream) {
  (void)in_sizes; (void)n_in; (void)out_size; (void)ws_size;
  const float* p  = (const float*)d_in[0];
  const float* Wh = (const float*)d_in[1];
  const float* bh = (const float*)d_in[2];
  const float* Wl = (const float*)d_in[3];
  const float* bl = (const float*)d_in[4];
  const float* Wg = (const float*)d_in[5];
  const float* bg = (const float*)d_in[6];
  float* out = (float*)d_out;

  char* ws = (char*)d_ws;
  unsigned short* h16 = (unsigned short*)(ws + 0);         // [B*N][64]   2 MB
  unsigned short* l16 = (unsigned short*)(ws + 2097152);   // [B*N][64]   2 MB
  unsigned short* g16 = (unsigned short*)(ws + 4194304);   // [B][C][N]   8 MB
  unsigned short* WhT = (unsigned short*)(ws + 12582912);  // [64][256]
  unsigned short* WlT = (unsigned short*)(ws + 12615680);  // [64][256]
  unsigned short* WgT = (unsigned short*)(ws + 12648448);  // [256][256]

  sa_wconv_kernel<<<(256 * 64 + 255) / 256, 256, 0, stream>>>(Wh, WhT, 256, 64);
  sa_wconv_kernel<<<(256 * 64 + 255) / 256, 256, 0, stream>>>(Wl, WlT, 256, 64);
  sa_wconv_kernel<<<(256 * 256 + 255) / 256, 256, 0, stream>>>(Wg, WgT, 256, 256);
  sa_proj_kernel<<<kB * kN / 64, 128, 0, stream>>>(p, WhT, WlT, WgT, bh, bl, bg,
                                                   h16, l16, g16);
  sa_attn_kernel<<<kB * kN / 64, 128, 0, stream>>>(p, h16, l16, g16, out);
}
